// Decoder_40544491274920
// MI455X (gfx1250) — matrix operand order repaired and checked by reading, not compiled, not run on hardware
//
#include <hip/hip_runtime.h>
#include <hip/hip_bf16.h>

// ---------------------------------------------------------------------------
// LSTM decoder w/ attention on MI455X (gfx1250).
// Strategy: f16 weights cached in workspace (fit in 192MB L2), all GEMMs via
// v_wmma_f32_16x16x32_f16 (wave32), f32 state/accumulation, per-step kernel
// chain on `stream` (graph-capture safe: no mallocs, no syncs).
// ---------------------------------------------------------------------------

#define B_  64
#define S_  64
#define T_  64
#define H_  1024
#define L_  2
#define FH_ 4096   // 4*H

typedef __attribute__((ext_vector_type(16))) _Float16 v16h;
typedef __attribute__((ext_vector_type(8)))  _Float16 v8h;
typedef __attribute__((ext_vector_type(8)))  float    v8f;

union frag16 { v16h v; v8h h[2]; };

// A-matrix fragment, 16x32 f16, row-major A[M][lda].
// ISA 7.12.2: lane m=lane&15; half=lane>>4; elements 0..7 = K(half*8..+7),
// elements 8..15 = K(half*8+16..+23).  Two 16B contiguous loads.
__device__ __forceinline__ v16h load_a_frag(const _Float16* __restrict__ A,
                                            int lda, int m0, int k0, int lane) {
  const int row  = m0 + (lane & 15);
  const int half = lane >> 4;
  const _Float16* p = A + (size_t)row * lda + k0 + half * 8;
  frag16 f;
  f.h[0] = *(const v8h*)(p);
  f.h[1] = *(const v8h*)(p + 16);
  return f.v;
}

// B-matrix fragment, 32x16 f16 (KxN), sourced from W stored row-major [N][ldw]
// (i.e. B = W^T, so B column n == W row n).  Lane n=lane&15; half=lane>>4;
// elements 0..7 = K(half*8..+7), elements 8..15 = K(half*8+16..+23).
__device__ __forceinline__ v16h load_b_frag(const _Float16* __restrict__ W,
                                            int ldw, int n0, int k0, int lane) {
  const int row  = n0 + (lane & 15);
  const int half = lane >> 4;
  const _Float16* p = W + (size_t)row * ldw + k0 + half * 8;
  frag16 f;
  f.h[0] = *(const v8h*)(p);
  f.h[1] = *(const v8h*)(p + 16);
  return f.v;
}

// D[64,N] = A1[64,K] @ W1[N,:]^T (+ A2[64,K] @ W2[N,:]^T) (+ tanh epilogue)
// One wave computes a 16x64 output strip: A frag reused across 4 B tiles.
// grid = (4 * N/64) / 8 blocks of 256 threads (8 waves).
__global__ void gemm_dual_wmma(const _Float16* __restrict__ A1,
                               const _Float16* __restrict__ W1, int ldw1,
                               const _Float16* __restrict__ A2,
                               const _Float16* __restrict__ W2, int ldw2,
                               float* __restrict__ D, int N, int K,
                               int apply_tanh) {
  const int lane    = threadIdx.x & 31;
  const int wave    = (blockIdx.x * blockDim.x + threadIdx.x) >> 5;
  const int nstrips = N >> 6;               // strips of 64 columns
  const int m0      = (wave / nstrips) * 16;
  const int n0      = (wave % nstrips) * 64;

  v8f acc0 = {}, acc1 = {}, acc2 = {}, acc3 = {};

  for (int k0 = 0; k0 < K; k0 += 32) {
    v16h a  = load_a_frag(A1, K, m0, k0, lane);
    v16h b0 = load_b_frag(W1, ldw1, n0 +  0, k0, lane);
    v16h b1 = load_b_frag(W1, ldw1, n0 + 16, k0, lane);
    v16h b2 = load_b_frag(W1, ldw1, n0 + 32, k0, lane);
    v16h b3 = load_b_frag(W1, ldw1, n0 + 48, k0, lane);
    acc0 = __builtin_amdgcn_wmma_f32_16x16x32_f16(false, a, false, b0, (short)0, acc0, false, false);
    acc1 = __builtin_amdgcn_wmma_f32_16x16x32_f16(false, a, false, b1, (short)0, acc1, false, false);
    acc2 = __builtin_amdgcn_wmma_f32_16x16x32_f16(false, a, false, b2, (short)0, acc2, false, false);
    acc3 = __builtin_amdgcn_wmma_f32_16x16x32_f16(false, a, false, b3, (short)0, acc3, false, false);
  }
  if (A2) {
    for (int k0 = 0; k0 < K; k0 += 32) {
      v16h a  = load_a_frag(A2, K, m0, k0, lane);
      v16h b0 = load_b_frag(W2, ldw2, n0 +  0, k0, lane);
      v16h b1 = load_b_frag(W2, ldw2, n0 + 16, k0, lane);
      v16h b2 = load_b_frag(W2, ldw2, n0 + 32, k0, lane);
      v16h b3 = load_b_frag(W2, ldw2, n0 + 48, k0, lane);
      acc0 = __builtin_amdgcn_wmma_f32_16x16x32_f16(false, a, false, b0, (short)0, acc0, false, false);
      acc1 = __builtin_amdgcn_wmma_f32_16x16x32_f16(false, a, false, b1, (short)0, acc1, false, false);
      acc2 = __builtin_amdgcn_wmma_f32_16x16x32_f16(false, a, false, b2, (short)0, acc2, false, false);
      acc3 = __builtin_amdgcn_wmma_f32_16x16x32_f16(false, a, false, b3, (short)0, acc3, false, false);
    }
  }

  // C/D layout (ISA 7.12.2): element r -> row = r + 8*(lane>>4), col = lane&15
  const int half = lane >> 4;
  const int col  = lane & 15;
  #pragma unroll
  for (int r = 0; r < 8; ++r) {
    const int row = m0 + r + 8 * half;
    float v0 = acc0[r], v1 = acc1[r], v2 = acc2[r], v3 = acc3[r];
    if (apply_tanh) { v0 = tanhf(v0); v1 = tanhf(v1); v2 = tanhf(v2); v3 = tanhf(v3); }
    float* dr = D + (size_t)row * N + n0 + col;
    dr[0]  = v0;  dr[16] = v1;  dr[32] = v2;  dr[48] = v3;
  }
}

__global__ void cvt_f32_f16(const float* __restrict__ s, _Float16* __restrict__ d, int n) {
  for (int i = blockIdx.x * blockDim.x + threadIdx.x; i < n;
       i += gridDim.x * blockDim.x)
    d[i] = (_Float16)s[i];
}

__global__ void embed_gather(const int* __restrict__ ids,
                             const float* __restrict__ emb,
                             _Float16* __restrict__ x, int t) {
  const int idx = blockIdx.x * blockDim.x + threadIdx.x;   // [0, B*H)
  const int b = idx >> 10, e = idx & 1023;
  const int tok = ids[t * B_ + b];
  x[idx] = (_Float16)emb[(size_t)tok * H_ + e];
}

__device__ __forceinline__ float sigf(float x) { return 1.0f / (1.0f + __expf(-x)); }

// gate order i,f,g,o (torch LSTMCell). Writes f32 state + f16 copy of h for
// next-layer / recurrent GEMM input.
__global__ void lstm_cell(const float* __restrict__ gates,
                          const float* __restrict__ b_ih,
                          const float* __restrict__ b_hh,
                          float* __restrict__ h_state,
                          float* __restrict__ c_state,
                          _Float16* __restrict__ h_f16) {
  const int idx = blockIdx.x * blockDim.x + threadIdx.x;   // [0, B*H)
  const int b = idx >> 10, j = idx & 1023;
  const float* g = gates + (size_t)b * FH_;
  const float gi = g[j]            + b_ih[j]            + b_hh[j];
  const float gf = g[H_ + j]       + b_ih[H_ + j]       + b_hh[H_ + j];
  const float gg = g[2 * H_ + j]   + b_ih[2 * H_ + j]   + b_hh[2 * H_ + j];
  const float go = g[3 * H_ + j]   + b_ih[3 * H_ + j]   + b_hh[3 * H_ + j];
  const float c1 = sigf(gf) * c_state[idx] + sigf(gi) * tanhf(gg);
  const float h1 = sigf(go) * tanhf(c1);
  c_state[idx] = c1;
  h_state[idx] = h1;
  h_f16[idx]   = (_Float16)h1;
}

// scores[b,s] = dot(qproj[b,:], ctx[s,b,:]); one wave per (b,s) pair.
__global__ void attn_scores(const float* __restrict__ qproj,
                            const float* __restrict__ ctx,
                            float* __restrict__ scores) {
  const int gid  = blockIdx.x * blockDim.x + threadIdx.x;
  const int wave = gid >> 5, lane = gid & 31;
  const int b = wave >> 6, s = wave & 63;
  const float* q = qproj + (size_t)b * H_;
  const float* c = ctx + (size_t)s * B_ * H_ + (size_t)b * H_;
  float sum = 0.0f;
  for (int h = lane; h < H_; h += 32) sum += q[h] * c[h];
  for (int m = 16; m; m >>= 1) sum += __shfl_xor(sum, m, 32);
  if (lane == 0) scores[b * S_ + s] = sum;
}

__global__ void softmax64(const float* __restrict__ scores, float* __restrict__ attn) {
  __shared__ float buf[S_];
  const int b = blockIdx.x, s = threadIdx.x;
  const float v = scores[b * S_ + s];
  buf[s] = v; __syncthreads();
  float mx = buf[0];
  for (int i = 1; i < S_; ++i) mx = fmaxf(mx, buf[i]);
  const float e = __expf(v - mx);
  __syncthreads();
  buf[s] = e; __syncthreads();
  float sum = 0.0f;
  for (int i = 0; i < S_; ++i) sum += buf[i];
  attn[b * S_ + s] = e / sum;
}

__global__ void attn_weighted(const float* __restrict__ attn,
                              const float* __restrict__ ctx,
                              _Float16* __restrict__ wtd) {
  const int idx = blockIdx.x * blockDim.x + threadIdx.x;   // [0, B*H)
  const int b = idx >> 10, h = idx & 1023;
  const float* a = attn + b * S_;
  float sum = 0.0f;
  for (int s = 0; s < S_; ++s)
    sum += a[s] * ctx[(size_t)s * B_ * H_ + (size_t)b * H_ + h];
  wtd[idx] = (_Float16)sum;
}

static inline char* ws_carve(char*& p, size_t bytes) {
  char* r = p;
  p += (bytes + 255) & ~(size_t)255;
  return r;
}

extern "C" void kernel_launch(void* const* d_in, const int* in_sizes, int n_in,
                              void* d_out, int out_size, void* d_ws, size_t ws_size,
                              hipStream_t stream) {
  const int*   ids   = (const int*)d_in[0];     // [T,B]
  const float* ctx   = (const float*)d_in[1];   // [S,B,H]
  // d_in[2] init_output: unused (input_feed=False)
  const float* h0    = (const float*)d_in[3];   // [L,B,H]
  const float* c0    = (const float*)d_in[4];
  const float* emb   = (const float*)d_in[5];   // [V,E]
  const float* W_ih  = (const float*)d_in[6];   // [L,4H,H]
  const float* W_hh  = (const float*)d_in[7];
  const float* b_ih  = (const float*)d_in[8];   // [L,4H]
  const float* b_hh  = (const float*)d_in[9];
  const float* W_in  = (const float*)d_in[10];  // [H,H]
  const float* W_out = (const float*)d_in[11];  // [H,2H]
  float* out = (float*)d_out;

  // ---- workspace layout (~43 MB) ----
  char* p = (char*)d_ws;
  _Float16* Wih16  = (_Float16*)ws_carve(p, (size_t)L_ * FH_ * H_ * 2);
  _Float16* Whh16  = (_Float16*)ws_carve(p, (size_t)L_ * FH_ * H_ * 2);
  _Float16* Win16  = (_Float16*)ws_carve(p, (size_t)H_ * H_ * 2);
  _Float16* Wout16 = (_Float16*)ws_carve(p, (size_t)H_ * 2 * H_ * 2);
  float*    h_st   = (float*)   ws_carve(p, (size_t)L_ * B_ * H_ * 4);
  float*    c_st   = (float*)   ws_carve(p, (size_t)L_ * B_ * H_ * 4);
  _Float16* hf16   = (_Float16*)ws_carve(p, (size_t)L_ * B_ * H_ * 2);
  _Float16* x16    = (_Float16*)ws_carve(p, (size_t)B_ * H_ * 2);
  float*    gates  = (float*)   ws_carve(p, (size_t)B_ * FH_ * 4);
  float*    qproj  = (float*)   ws_carve(p, (size_t)B_ * H_ * 4);
  float*    scores = (float*)   ws_carve(p, (size_t)B_ * S_ * 4);
  float*    attn   = (float*)   ws_carve(p, (size_t)B_ * S_ * 4);
  _Float16* wtd16  = (_Float16*)ws_carve(p, (size_t)B_ * H_ * 2);

  // ---- one-time (per call): weights -> f16, init state ----
  auto cvt = [&](const float* s, _Float16* d, int n) {
    int blocks = (n + 255) / 256;
    if (blocks > 4096) blocks = 4096;
    cvt_f32_f16<<<blocks, 256, 0, stream>>>(s, d, n);
  };
  cvt(W_ih,  Wih16,  L_ * FH_ * H_);
  cvt(W_hh,  Whh16,  L_ * FH_ * H_);
  cvt(W_in,  Win16,  H_ * H_);
  cvt(W_out, Wout16, H_ * 2 * H_);
  cvt(h0,    hf16,   L_ * B_ * H_);
  hipMemcpyAsync(h_st, h0, (size_t)L_ * B_ * H_ * 4, hipMemcpyDeviceToDevice, stream);
  hipMemcpyAsync(c_st, c0, (size_t)L_ * B_ * H_ * 4, hipMemcpyDeviceToDevice, stream);

  const int pw_blocks = (B_ * H_) / 256;            // 256
  const int gemm_big  = FH_ / 128;                  // 32 blocks (256 waves)
  const int gemm_sm   = H_ / 128;                   // 8 blocks  (64 waves)

  for (int t = 0; t < T_; ++t) {
    embed_gather<<<pw_blocks, 256, 0, stream>>>(ids, emb, x16, t);

    const _Float16* xin = x16;
    for (int l = 0; l < L_; ++l) {
      gemm_dual_wmma<<<gemm_big, 256, 0, stream>>>(
          xin,              Wih16 + (size_t)l * FH_ * H_, H_,
          hf16 + l * B_ * H_, Whh16 + (size_t)l * FH_ * H_, H_,
          gates, FH_, H_, 0);
      lstm_cell<<<pw_blocks, 256, 0, stream>>>(
          gates, b_ih + l * FH_, b_hh + l * FH_,
          h_st + l * B_ * H_, c_st + l * B_ * H_, hf16 + l * B_ * H_);
      xin = hf16 + l * B_ * H_;                     // layer output (f16) feeds next
    }

    // attention: qproj = q @ W_in^T
    gemm_dual_wmma<<<gemm_sm, 256, 0, stream>>>(
        xin, Win16, H_, nullptr, nullptr, 0, qproj, H_, H_, 0);
    attn_scores<<<(B_ * S_ * 32) / 256, 256, 0, stream>>>(qproj, ctx, scores);
    softmax64<<<B_, S_, 0, stream>>>(scores, attn);
    attn_weighted<<<pw_blocks, 256, 0, stream>>>(attn, ctx, wtd16);

    // out_t = tanh([weighted, q] @ W_out^T); W_out row-major [H, 2H]:
    // cols 0..H-1 hit weighted, cols H..2H-1 hit q.
    gemm_dual_wmma<<<gemm_sm, 256, 0, stream>>>(
        wtd16, Wout16,       2 * H_,
        xin,   Wout16 + H_,  2 * H_,
        out + (size_t)t * B_ * H_, H_, H_, 1);
  }

  // tail: hT, cT, attns[-1]
  float* tail = out + (size_t)T_ * B_ * H_;
  hipMemcpyAsync(tail,                     h_st, (size_t)L_ * B_ * H_ * 4, hipMemcpyDeviceToDevice, stream);
  hipMemcpyAsync(tail + L_ * B_ * H_,      c_st, (size_t)L_ * B_ * H_ * 4, hipMemcpyDeviceToDevice, stream);
  hipMemcpyAsync(tail + 2 * L_ * B_ * H_,  attn, (size_t)B_ * S_ * 4,      hipMemcpyDeviceToDevice, stream);
}
